// Qwen3NextGatedDeltaNet_82540681494773
// MI455X (gfx1250) — matrix operand order repaired and checked by reading, not compiled, not run on hardware
//
#include <hip/hip_runtime.h>
#include <hip/hip_bf16.h>

// ---------------- problem constants (match reference) ----------------
#define HKn 16
#define HVn 32
#define DKn 128
#define DVn 128
#define KWn 4
#define Bn  2
#define Tn  2048
#define Cn  (HKn*DKn*2 + HVn*DVn)   // 8192
#define CH  64                       // chunk length for WY-form delta rule
#define NCH (Tn/CH)

typedef _Float16 f16;
typedef __attribute__((ext_vector_type(16))) _Float16 v16h;
typedef __attribute__((ext_vector_type(8)))  _Float16 v8h;
typedef __attribute__((ext_vector_type(8)))  float    v8f;

// D = A(16x32 f16) * B(32x16 f16) + C(16x16 f32)
#define WMMA(a,b,c) __builtin_amdgcn_wmma_f32_16x16x32_f16(false,(a),false,(b),(short)0,(c),false,false)

// ---- CDNA5 async global->LDS (ASYNCcnt-tracked, no VGPR round trip) ----
__device__ __forceinline__ void async_ld16(unsigned ldsOff, const void* gptr) {
  asm volatile("global_load_async_to_lds_b128 %0, %1, off"
               :: "v"(ldsOff), "v"(gptr) : "memory");
}
__device__ __forceinline__ void wait_async0() {
  asm volatile("s_wait_asynccnt 0" ::: "memory");
}

// ---- CDNA5 ISA 7.12.2 fragment index maps (wave32, 16-bit, 16x16x32) ----
// A: lane owns row M=(lane&15); element e holds K = (e&7) + 8*(lane>>4) + 16*(e>>3)
__device__ __forceinline__ int kmapA(int e, int hf) { return (e & 7) + (hf << 3) + ((e >> 3) << 4); }
// B: lane owns col N=(lane&15); element e holds K = (e&7) + 8*(lane>>4) + 16*(e>>3)
__device__ __forceinline__ int kmapB(int e, int hf) { return (e & 7) + (hf << 3) + ((e >> 3) << 4); }

// A operand from row-major f16 [rows][ld] (contiguous per half-lane -> ds_load_b128)
__device__ __forceinline__ v16h fragA_f16(const f16* src, int ld, int mrow, int kb, int lane) {
  v16h a; int r = mrow + (lane & 15); int hf = lane >> 4;
#pragma unroll
  for (int e = 0; e < 16; ++e) a[e] = src[r*ld + kb + kmapA(e, hf)];
  return a;
}
// A operand with per-row decay scale (gamma_i <= 1, f16-safe)
__device__ __forceinline__ v16h fragA_f16_scaled(const f16* src, int ld, int mrow, int kb, int lane, float s) {
  v16h a; int r = mrow + (lane & 15); int hf = lane >> 4;
#pragma unroll
  for (int e = 0; e < 16; ++e) a[e] = (f16)((float)src[r*ld + kb + kmapA(e, hf)] * s);
  return a;
}
// B operand representing (row-major src)^T : B[k][n] = src[n][k] (contiguous reads)
__device__ __forceinline__ v16h fragBT_f16(const f16* src, int ld, int ncol, int kb, int lane) {
  v16h bf; int n = ncol + (lane & 15); int hf = lane >> 4;
#pragma unroll
  for (int e = 0; e < 16; ++e) bf[e] = src[n*ld + kb + kmapB(e, hf)];
  return bf;
}
// 16x16 f32 C/D tile <-> LDS (ISA layout: VGPR r = row r + 8*(lane>>4), col = lane&15)
__device__ __forceinline__ v8f ldC(const float* src, int ld, int mrow, int ncol, int lane, float s) {
  v8f c; int n = ncol + (lane & 15); int m0 = mrow + ((lane >> 4) << 3);
#pragma unroll
  for (int r = 0; r < 8; ++r) c[r] = src[(m0 + r)*ld + n] * s;
  return c;
}
__device__ __forceinline__ void stC(float* dst, int ld, int mrow, int ncol, int lane, v8f c) {
  int n = ncol + (lane & 15); int m0 = mrow + ((lane >> 4) << 3);
#pragma unroll
  for (int r = 0; r < 8; ++r) dst[(m0 + r)*ld + n] = c[r];
}
// state-update tile store: f32 S row-major + f16 S^T copy (vectorized b128 store)
__device__ __forceinline__ void stC_S(float* S_, f16* ST_, int mrow, int ncol, int lane, v8f c) {
  int n = ncol + (lane & 15); int m0 = mrow + ((lane >> 4) << 3);
  v8h t;
#pragma unroll
  for (int r = 0; r < 8; ++r) { S_[(m0 + r)*DVn + n] = c[r]; t[r] = (f16)c[r]; }
  *(v8h*)(ST_ + n*DKn + m0) = t;   // 16B aligned: m0 multiple of 8
}

// ============ kernel 1: depthwise causal conv + SiLU + l2norm + pack ============
__global__ __launch_bounds__(128) void gdn_prep_kernel(
    const float* __restrict__ x, const float* __restrict__ cw,
    f16* __restrict__ qw, f16* __restrict__ kw, float* __restrict__ vw)
{
  __shared__ float red[128];
  const int tid = threadIdx.x;
  const int blk = blockIdx.x;
  const int s = blk & 63;               // 64 channel segments of 128
  const int t = (blk >> 6) & (Tn - 1);
  const int b = blk >> 17;
  const int c = s*128 + tid;

  float acc = 0.f;
#pragma unroll
  for (int j = 0; j < KWn; ++j) {
    int tt = t - (KWn - 1) + j;
    float xv = (tt >= 0) ? x[((size_t)b*Tn + tt)*Cn + c] : 0.f;
    acc += xv * cw[c*KWn + j];
  }
  float y = acc / (1.f + __expf(-acc));            // SiLU

  if (s < 32) {                                    // q or k head: l2norm over 128 dims
    red[tid] = y*y;
    __syncthreads();
    for (int off = 64; off > 0; off >>= 1) {
      if (tid < off) red[tid] += red[tid + off];
      __syncthreads();
    }
    float r = rsqrtf(red[0] + 1e-6f);
    if (s < 16) {
      qw[(((size_t)b*Tn + t)*HKn + s)*DKn + tid] = (f16)(y * r * 0.08838834764831845f); // DK^-0.5 folded
    } else {
      kw[(((size_t)b*Tn + t)*HKn + (s-16))*DKn + tid] = (f16)(y * r);
    }
  } else {                                         // v head
    vw[(((size_t)b*Tn + t)*HVn + (s-32))*DVn + tid] = y;
  }
}

// ============ kernel 2: beta = sigmoid(b), g = -exp(A_log)*softplus(a+dt_bias) ============
__global__ __launch_bounds__(256) void gdn_gate_kernel(
    const float* __restrict__ b_in, const float* __restrict__ a_in,
    const float* __restrict__ dtb, const float* __restrict__ alog,
    float* __restrict__ betw, float* __restrict__ gw, int n)
{
  int i = blockIdx.x*blockDim.x + threadIdx.x;
  if (i >= n) return;
  int h = i % HVn;
  betw[i] = 1.f / (1.f + __expf(-b_in[i]));
  float xv = a_in[i] + dtb[h];
  float sp = (xv > 20.f) ? xv : log1pf(__expf(xv));
  gw[i] = -__expf(alog[h]) * sp;                   // always <= 0 -> all decay factors <= 1
}

// ============ kernel 3: chunked WY-form gated delta-rule scan (WMMA) ============
// Dynamic LDS layout (303,616 B — needs CDNA5's 320KB/WGP LDS):
//   S   f32 128x128 @      0   persistent state
//   O   f32  64x128 @  65536   output accumulator
//   Dlt f32  64x128 @  98304   KV0, then delta
//   KK  f32  64x64  @ 131072   raw K K^T, then TA
//   QK  f32  64x64  @ 147456   raw Q K^T
//   bcum/betc        @ 163840 / 164096
//   Kc0 f16  64x128 @ 164352   \  double-buffered async-staged
//   Qc0 f16  64x128 @ 180736    | chunk operands
//   DhT f16 128x64  @ 197120   Delta^T   (B operands -> contiguous)
//   ST  f16 128x128 @ 213504   S^T       (B operands -> contiguous)
//   Wh  f16  64x64  @ 246272   masked/decayed QK^T (A operand, exp hoisted)
//   KhT f16 128x64  @ 254464   decayed K^T (A operand, exp hoisted)
//   Kc1 f16  64x128 @ 270848   /
//   Qc1 f16  64x128 @ 287232  /
#define KC0_OFF 164352u
#define QC0_OFF 180736u
#define KC1_OFF 270848u
#define QC1_OFF 287232u
#define SMEM_BYTES 303616

// issue one chunk's K/Q staging as async 16B packets (tracked by ASYNCcnt)
__device__ __forceinline__ void stage_async(const f16* kw, const f16* qw,
    int b, int t0, int kh, unsigned kcOff, unsigned qcOff, int tid) {
  for (int idx = tid; idx < (CH*DKn)/8; idx += 256) {   // 1024 packets of 8 halves
    int i = idx >> 4, d = (idx & 15) << 3;
    size_t base = (((size_t)b*Tn + t0 + i)*HKn + kh)*DKn + d;
    async_ld16(kcOff + (unsigned)idx*16u, kw + base);
    async_ld16(qcOff + (unsigned)idx*16u, qw + base);
  }
}

__global__ __launch_bounds__(256) void gdn_scan_kernel(
    const f16* __restrict__ qw, const f16* __restrict__ kw,
    const float* __restrict__ vw, const float* __restrict__ betw,
    const float* __restrict__ gw, float* __restrict__ out)
{
  extern __shared__ char smem[];
  float* S    = (float*)(smem + 0);
  float* O    = (float*)(smem + 65536);
  float* Dlt  = (float*)(smem + 98304);
  float* KK   = (float*)(smem + 131072);
  float* QK   = (float*)(smem + 147456);
  float* bcum = (float*)(smem + 163840);
  float* betc = (float*)(smem + 164096);
  f16*   DhT  = (f16*)(smem + 197120);
  f16*   ST   = (f16*)(smem + 213504);
  f16*   Wh   = (f16*)(smem + 246272);
  f16*   KhT  = (f16*)(smem + 254464);
  const unsigned kcOffs[2] = {KC0_OFF, KC1_OFF};
  const unsigned qcOffs[2] = {QC0_OFF, QC1_OFF};

  const int tid  = threadIdx.x;
  const int wv   = tid >> 5, lane = tid & 31;      // wave32
  const int b    = blockIdx.x / HVn;
  const int h    = blockIdx.x % HVn;
  const int kh   = h >> 1;                          // GQA: HV/HK = 2

  for (int i = tid; i < DKn*DVn; i += 256) { S[i] = 0.f; ST[i] = (f16)0.f; }

  // preload chunk 0 into buffer 0 (async, overlapped with S init)
  stage_async(kw, qw, b, 0, kh, KC0_OFF, QC0_OFF, tid);

  for (int ch = 0; ch < NCH; ++ch) {
    const int t0 = ch * CH;
    const int cur = ch & 1;
    const f16* Kc = (const f16*)(smem + kcOffs[cur]);
    const f16* Qc = (const f16*)(smem + qcOffs[cur]);

    // retire this chunk's async staging, publish LDS to all waves
    wait_async0();
    __syncthreads();
    // issue next chunk's staging into the other buffer; overlaps all compute below
    if (ch + 1 < NCH)
      stage_async(kw, qw, b, t0 + CH, kh, kcOffs[cur ^ 1], qcOffs[cur ^ 1], tid);

    // ---- gates for this chunk; prefetch next chunk's v rows ----
    if (tid < CH) {
      size_t gi = ((size_t)b*Tn + t0 + tid)*HVn + h;
      betc[tid] = betw[gi];
      bcum[tid] = gw[gi];
      if (ch + 1 < NCH) {
        const float* vp = &vw[(((size_t)b*Tn + t0 + CH + tid)*HVn + h)*DVn];
        __builtin_prefetch(vp, 0, 1);
        __builtin_prefetch(vp + 64, 0, 1);
      }
    }
    __syncthreads();
    if (tid == 0) {                                 // cumulative log-decay b_i
      float s = 0.f;
      for (int i = 0; i < CH; ++i) { s += bcum[i]; bcum[i] = s; }
    }
    __syncthreads();

    // ---- phase 1: KK = K K^T, QK = Q K^T (64x64); KV0/QS0 = diag(gamma) {K,Q} @ S ----
    for (int tl = wv; tl < 16; tl += 8) {
      int mt = (tl >> 2) << 4, nt = (tl & 3) << 4;
      v8f ck = {}, cq = {};
#pragma unroll
      for (int kb = 0; kb < DKn; kb += 32) {
        v16h bb = fragBT_f16(Kc, DKn, nt, kb, lane);
        v16h ak = fragA_f16(Kc, DKn, mt, kb, lane);
        ck = WMMA(ak, bb, ck);
        v16h aq = fragA_f16(Qc, DKn, mt, kb, lane);
        cq = WMMA(aq, bb, cq);
      }
      stC(KK, CH, mt, nt, lane, ck);
      stC(QK, CH, mt, nt, lane, cq);
    }
    for (int tl = wv; tl < 32; tl += 8) {
      int mt = (tl >> 3) << 4, nt = (tl & 7) << 4;
      float rs = __expf(bcum[mt + (lane & 15)]);    // gamma_i <= 1
      v8f ck = {}, cq = {};
#pragma unroll
      for (int kb = 0; kb < DKn; kb += 32) {
        v16h bb = fragBT_f16(ST, DKn, nt, kb, lane);          // B = S (via S^T copy)
        v16h ak = fragA_f16_scaled(Kc, DKn, mt, kb, lane, rs);
        ck = WMMA(ak, bb, ck);
        v16h aq = fragA_f16_scaled(Qc, DKn, mt, kb, lane, rs);
        cq = WMMA(aq, bb, cq);
      }
      stC(Dlt, DVn, mt, nt, lane, ck);              // KV0
      stC(O,   DVn, mt, nt, lane, cq);              // QS0 (output base term)
    }
    __syncthreads();

    // ---- build TA (in-place on KK), Wh = mask/decay(QK), RHS = beta*(v-KV0) ----
    for (int idx = tid; idx < CH*CH; idx += 256) {
      int i = idx >> 6, j = idx & 63;
      float dec = __expf(bcum[i] - bcum[j]);
      KK[idx] = (j < i)  ? betc[i] * dec * KK[idx] : 0.f;
      Wh[idx] = (j <= i) ? (f16)(QK[idx] * dec)    : (f16)0.f;
    }
    for (int idx = tid; idx < CH*DVn; idx += 256) {
      int i = idx >> 7;
      float vv = vw[(((size_t)b*Tn + t0 + i)*HVn + h)*DVn + (idx & 127)];
      Dlt[idx] = betc[i] * (vv - Dlt[idx]);
    }
    __syncthreads();

    // ---- forward substitution: delta_i = RHS_i - sum_{j<i} TA_{ij} delta_j ----
    for (int i = 1; i < CH; ++i) {
      if (tid < DVn) {
        float acc = Dlt[i*DVn + tid];
        for (int j = 0; j < i; ++j)
          acc -= KK[i*CH + j] * Dlt[j*DVn + tid];
        Dlt[i*DVn + tid] = acc;
      }
      __syncthreads();
    }

    // ---- build f16 transposed operands: DhT = Delta^T, KhT = decayed K^T ----
    const float bL = bcum[CH-1];
    for (int idx = tid; idx < CH*DVn; idx += 256) {
      int i = idx >> 7, d = idx & 127;
      DhT[d*CH + i] = (f16)Dlt[idx];
    }
    for (int idx = tid; idx < CH*DKn; idx += 256) {
      int dk = idx >> 6, j = idx & 63;              // contiguous KhT writes
      KhT[idx] = (f16)((float)Kc[j*DKn + dk] * __expf(bL - bcum[j]));
    }
    __syncthreads();

    // ---- phase 2: O += Wh @ Delta ; S = gL*S + KhT @ Delta (also refresh ST) ----
    const float gL = __expf(bL);
    for (int tl = wv; tl < 32; tl += 8) {
      int mt = (tl >> 3) << 4, nt = (tl & 7) << 4;
      v8f c = ldC(O, DVn, mt, nt, lane, 1.f);
#pragma unroll
      for (int kb = 0; kb < CH; kb += 32) {
        v16h a  = fragA_f16(Wh, CH, mt, kb, lane);
        v16h bb = fragBT_f16(DhT, CH, nt, kb, lane);
        c = WMMA(a, bb, c);
      }
      stC(O, DVn, mt, nt, lane, c);
    }
    for (int tl = wv; tl < 64; tl += 8) {
      int mt = (tl >> 3) << 4, nt = (tl & 7) << 4;
      v8f c = ldC(S, DVn, mt, nt, lane, gL);
#pragma unroll
      for (int kb = 0; kb < CH; kb += 32) {
        v16h a  = fragA_f16(KhT, CH, mt, kb, lane);
        v16h bb = fragBT_f16(DhT, CH, nt, kb, lane);
        c = WMMA(a, bb, c);
      }
      stC_S(S, ST, mt, nt, lane, c);
    }
    __syncthreads();

    // ---- write chunk output: out[b, t0+i, h*DV + dv] ----
    for (int idx = tid; idx < CH*DVn; idx += 256) {
      int i = idx >> 7, d = idx & 127;
      out[(((size_t)b*Tn + t0 + i)*HVn + h)*DVn + d] = O[idx];
    }
    __syncthreads();
  }
}

// =============================== host launcher ===============================
extern "C" void kernel_launch(void* const* d_in, const int* in_sizes, int n_in,
                              void* d_out, int out_size, void* d_ws, size_t ws_size,
                              hipStream_t stream) {
  const float* x    = (const float*)d_in[0];   // mixed_qkv [B,T,C]
  const float* b_in = (const float*)d_in[1];   // b [B,T,HV]
  const float* a_in = (const float*)d_in[2];   // a [B,T,HV]
  const float* cw   = (const float*)d_in[3];   // conv_weights [C,KW]
  const float* dtb  = (const float*)d_in[4];   // dt_bias [HV]
  const float* alog = (const float*)d_in[5];   // A_log [HV]
  float* out = (float*)d_out;

  // workspace partition (~102 MB)
  char* ws = (char*)d_ws;
  size_t off = 0;
  f16*   qw   = (f16*)(ws + off); off += (size_t)Bn*Tn*HKn*DKn*sizeof(f16);
  f16*   kw   = (f16*)(ws + off); off += (size_t)Bn*Tn*HKn*DKn*sizeof(f16);
  float* vw   = (float*)(ws + off); off += (size_t)Bn*Tn*HVn*DVn*sizeof(float);
  float* betw = (float*)(ws + off); off += (size_t)Bn*Tn*HVn*sizeof(float);
  float* gw   = (float*)(ws + off); off += (size_t)Bn*Tn*HVn*sizeof(float);
  (void)off; (void)ws_size; (void)in_sizes; (void)n_in; (void)out_size;

  // opt in to >64KB dynamic LDS for the scan kernel (CDNA5: 320KB per WGP)
  hipFuncSetAttribute(reinterpret_cast<const void*>(gdn_scan_kernel),
                      hipFuncAttributeMaxDynamicSharedMemorySize, SMEM_BYTES);

  gdn_prep_kernel<<<Bn*Tn*(Cn/128), 128, 0, stream>>>(x, cw, qw, kw, vw);

  int ng = Bn*Tn*HVn;
  gdn_gate_kernel<<<(ng + 255)/256, 256, 0, stream>>>(b_in, a_in, dtb, alog, betw, gw, ng);

  gdn_scan_kernel<<<Bn*HVn, 256, SMEM_BYTES, stream>>>(qw, kw, vw, betw, gw, out);
}
